// MyGat_14671608283641
// MI455X (gfx1250) — matrix operand order repaired and checked by reading, not compiled, not run on hardware
//
#include <hip/hip_runtime.h>
#include <hip/hip_bf16.h>

// GAT layer, MI455X (gfx1250).
// Roofline: adj (268MB) read exactly once, coalesced 128B per row per WG ->
// ~11.5us floor at 23.3 TB/s. exp(leakyrelu(a+b)) factored via monotonicity:
//   S[i,j] = adj[j,i] * max(ed[i]*es[j], ed2[i]*es2[j])
// -> single fused pass: bf16 WMMA aggregation + register rowsums, normalize
// in epilogue. H^T tile staging uses GLOBAL_LOAD_ASYNC_TO_LDS_B128 (ASYNCcnt);
// adj stream is software-pipelined with double-buffered LDS (one barrier per
// chunk). Workspace: 4MB (H f32) + 2MB (H^T bf16) + 128KB (exp tables).

#define N_NODES 8192
#define DIM     128

typedef __bf16 bf16_t;
typedef bf16_t bf16x16 __attribute__((ext_vector_type(16)));
typedef bf16_t bf16x8  __attribute__((ext_vector_type(8)));
typedef float  f32x8   __attribute__((ext_vector_type(8)));
typedef float  f32x4   __attribute__((ext_vector_type(4)));

#if defined(__has_builtin)
#if __has_builtin(__builtin_amdgcn_global_load_async_to_lds_b128)
#define HAVE_ASYNC_LDS 1
#endif
#endif
#ifndef HAVE_ASYNC_LDS
#define HAVE_ASYNC_LDS 0
#endif

#if HAVE_ASYNC_LDS
// Builtin signature (from clang diagnostic): params are
// "int __attribute__((vector_size(16)))*" in AS1 (global src) / AS3 (LDS dst).
typedef int v4i_t __attribute__((vector_size(16)));
typedef __attribute__((address_space(1))) v4i_t gv4i_t;
typedef __attribute__((address_space(3))) v4i_t lv4i_t;
#define ASYNC_B128(g, l, off) \
    __builtin_amdgcn_global_load_async_to_lds_b128((gv4i_t*)(g), (lv4i_t*)(l), (off), 0)
#if __has_builtin(__builtin_amdgcn_s_wait_asynccnt)
#define WAIT_ASYNC() __builtin_amdgcn_s_wait_asynccnt(0)
#else
#define WAIT_ASYNC() asm volatile("s_wait_asynccnt 0x0" ::: "memory")
#endif
#endif

// ---------------------------------------------------------------------------
// Kernel 1: H = x @ W^T  (16 rows per wave, 8 16-col tiles, WMMA bf16)
// Writes H (f32 row-major) and HbT (bf16, column-major: HbT[o*N + n]).
// ---------------------------------------------------------------------------
__global__ __launch_bounds__(32)
void gat_h_kernel(const float* __restrict__ x, const float* __restrict__ w,
                  float* __restrict__ H, bf16_t* __restrict__ HbT) {
    const int lane   = threadIdx.x & 31;
    const int l15    = lane & 15;
    const bool hiH   = lane >= 16;
    const int r0     = blockIdx.x * 16;

    f32x8 acc[8];
    for (int t = 0; t < 8; ++t)
        for (int e = 0; e < 8; ++e) acc[t][e] = 0.0f;

    for (int kc = 0; kc < DIM; kc += 32) {
        // A fragment: 16x32 bf16. lane<16: row=l15, K = kc+{0..7,16..23};
        // lane>=16: row=l15, K = kc+{8..15,24..31}.
        bf16x16 af;
        {
            const int row = r0 + l15;
            const int kb  = kc + (hiH ? 8 : 0);
            f32x4 a0 = *(const f32x4*)(x + row * DIM + kb);
            f32x4 a1 = *(const f32x4*)(x + row * DIM + kb + 4);
            f32x4 a2 = *(const f32x4*)(x + row * DIM + kb + 16);
            f32x4 a3 = *(const f32x4*)(x + row * DIM + kb + 20);
            for (int e = 0; e < 4; ++e) {
                af[e]      = (bf16_t)a0[e];
                af[4 + e]  = (bf16_t)a1[e];
                af[8 + e]  = (bf16_t)a2[e];
                af[12 + e] = (bf16_t)a3[e];
            }
        }
        for (int t = 0; t < 8; ++t) {
            // B fragment: 32x16 bf16, B[k][o] = W[o][k].
            bf16x16 bf;
            const int orow = t * 16 + l15;
            const int kb2  = kc + (hiH ? 8 : 0);
            f32x4 b0 = *(const f32x4*)(w + orow * DIM + kb2);
            f32x4 b1 = *(const f32x4*)(w + orow * DIM + kb2 + 4);
            f32x4 b2 = *(const f32x4*)(w + orow * DIM + kb2 + 16);
            f32x4 b3 = *(const f32x4*)(w + orow * DIM + kb2 + 20);
            for (int e = 0; e < 4; ++e) {
                bf[e]      = (bf16_t)b0[e];
                bf[4 + e]  = (bf16_t)b1[e];
                bf[8 + e]  = (bf16_t)b2[e];
                bf[12 + e] = (bf16_t)b3[e];
            }
            acc[t] = __builtin_amdgcn_wmma_f32_16x16x32_bf16(
                false, af, false, bf, (short)0, acc[t], false, false);
        }
    }
    // C layout: VGPR r, lane<16 -> M=r ; lane>=16 -> M=r+8 ; N = l15 (+tile).
    for (int t = 0; t < 8; ++t)
        for (int r = 0; r < 8; ++r) {
            const int M   = r + (hiH ? 8 : 0);
            const int col = t * 16 + l15;
            const int row = r0 + M;
            const float v = acc[t][r];
            H[row * DIM + col]               = v;
            HbT[(size_t)col * N_NODES + row] = (bf16_t)v;
        }
}

// ---------------------------------------------------------------------------
// Kernel 2: per-node attention logits -> 4 exp tables (factored leaky-relu).
// ---------------------------------------------------------------------------
__global__ __launch_bounds__(256)
void gat_alpha_kernel(const float* __restrict__ H,
                      const float* __restrict__ att_src,
                      const float* __restrict__ att_dst,
                      float* __restrict__ es, float* __restrict__ es2,
                      float* __restrict__ ed, float* __restrict__ ed2) {
    const int r = blockIdx.x * 256 + threadIdx.x;
    float as = 0.0f, ad = 0.0f;
    for (int c = 0; c < DIM; ++c) {
        const float h = H[r * DIM + c];
        as += h * att_src[c];
        ad += h * att_dst[c];
    }
    es[r]  = __expf(as);
    es2[r] = __expf(0.2f * as);
    ed[r]  = __expf(ad);
    ed2[r] = __expf(0.2f * ad);
}

// ---------------------------------------------------------------------------
// Kernel 3: single pass over adj. WG owns 32 output rows (i0..i0+31).
// Double-buffered LDS + register-pipelined adj stream + async H^T staging.
// One barrier per chunk (distance-2 buffer reuse -> intervening barrier).
// ---------------------------------------------------------------------------
#define SSTRIDE 40   // 80B rows -> every 16B fragment load aligned
#define NCHUNK  (N_NODES / 32)

__global__ __launch_bounds__(256)
void gat_aggregate_kernel(const float* __restrict__ adj,
                          const bf16_t* __restrict__ HbT,
                          const float* __restrict__ es,
                          const float* __restrict__ es2,
                          const float* __restrict__ ed,
                          const float* __restrict__ ed2,
                          float* __restrict__ Z) {
    __shared__ bf16_t sS[2][32 * SSTRIDE];    // S tile, [i_local][j_local]
    __shared__ bf16_t sHT[2][128 * SSTRIDE];  // H^T tile, [o][j_local]
    __shared__ float  sRS[32];

    const int t    = threadIdx.x;
    const int lane = t & 31;
    const int wave = t >> 5;
    const int l15  = lane & 15;
    const bool hiH = lane >= 16;
    const int i0   = blockIdx.x * 32;
    const int rh   = wave & 1;   // which 16-row half
    const int cg   = wave >> 1;  // which 32-col group (2 tiles)

    // adj-staging role: thread covers (j_local = t>>3, 4 cols at (t&7)*4)
    const int jj  = t >> 3;
    const int iiq = (t & 7) * 4;
    float edv[4], ed2v[4];
    for (int q = 0; q < 4; ++q) {
        edv[q]  = ed[i0 + iiq + q];
        ed2v[q] = ed2[i0 + iiq + q];
    }
    float rs[4] = {0.0f, 0.0f, 0.0f, 0.0f};

    // HbT-staging role: thread covers (o = t>>1, 16 j at (t&1)*16)
    const int oo = t >> 1;
    const int jh = (t & 1) * 16;
    const bf16_t* hsrc = HbT + (size_t)oo * N_NODES + jh;

    f32x8 acc0, acc1;
    for (int e = 0; e < 8; ++e) { acc0[e] = 0.0f; acc1[e] = 0.0f; }

    // ---- pipeline prologue: chunk 0 ----
    f32x4 a4 = *(const f32x4*)(adj + (size_t)jj * N_NODES + i0 + iiq);
#if HAVE_ASYNC_LDS
    {
        bf16_t* l = &sHT[0][oo * SSTRIDE + jh];
        ASYNC_B128(hsrc, l, 0);
        ASYNC_B128(hsrc, l, 16);
    }
#endif

    for (int n = 0; n < NCHUNK; ++n) {
        const int j0  = n * 32;
        const int buf = n & 1;

        // --- stage S tile for chunk n (fused scale + transpose) ---
        const f32x4 acur = a4;
        if (n + 1 < NCHUNK)  // register-prefetch next adj tile (issues early)
            a4 = *(const f32x4*)(adj + (size_t)(j0 + 32 + jj) * N_NODES + i0 + iiq);
        const float esj  = es[j0 + jj];
        const float es2j = es2[j0 + jj];
        for (int q = 0; q < 4; ++q) {
            const float sv = acur[q] * fmaxf(edv[q] * esj, ed2v[q] * es2j);
            rs[q] += sv;
            sS[buf][(iiq + q) * SSTRIDE + jj] = (bf16_t)sv;
        }
#if !HAVE_ASYNC_LDS
        // synchronous fallback: copy current H^T tile through VGPRs
        {
            const uint4* src = (const uint4*)(hsrc + j0);
            uint4* dst = (uint4*)(&sHT[buf][oo * SSTRIDE + jh]);
            dst[0] = src[0];
            dst[1] = src[1];
        }
#endif
        if (n + 2 < NCHUNK)
            __builtin_prefetch(adj + (size_t)(j0 + 64 + jj) * N_NODES + i0 + iiq, 0, 1);
#if HAVE_ASYNC_LDS
        WAIT_ASYNC();   // chunk n's H^T tile resident in sHT[buf]
#endif
        __syncthreads();

        // --- WMMA: A = S[16x32], B = H tile [32x16] x2 ---
        bf16x16 af;
        {
            const int iloc = rh * 16 + l15;
            const int c0   = hiH ? 16 : 0;
            bf16x8 lo = *(const bf16x8*)(&sS[buf][iloc * SSTRIDE + c0]);
            bf16x8 hi = *(const bf16x8*)(&sS[buf][iloc * SSTRIDE + c0 + 8]);
            for (int e = 0; e < 8; ++e) { af[e] = lo[e]; af[8 + e] = hi[e]; }
        }
        bf16x16 b0, b1;
        {
            const int off = hiH ? 16 : 0;
            bf16x8 p0 = *(const bf16x8*)(&sHT[buf][(cg * 32 + l15) * SSTRIDE + off]);
            bf16x8 p1 = *(const bf16x8*)(&sHT[buf][(cg * 32 + l15) * SSTRIDE + off + 8]);
            bf16x8 p2 = *(const bf16x8*)(&sHT[buf][(cg * 32 + 16 + l15) * SSTRIDE + off]);
            bf16x8 p3 = *(const bf16x8*)(&sHT[buf][(cg * 32 + 16 + l15) * SSTRIDE + off + 8]);
            for (int e = 0; e < 8; ++e) {
                b0[e] = p0[e]; b0[8 + e] = p1[e];
                b1[e] = p2[e]; b1[8 + e] = p3[e];
            }
        }
        acc0 = __builtin_amdgcn_wmma_f32_16x16x32_bf16(
            false, af, false, b0, (short)0, acc0, false, false);
        acc1 = __builtin_amdgcn_wmma_f32_16x16x32_bf16(
            false, af, false, b1, (short)0, acc1, false, false);

#if HAVE_ASYNC_LDS
        // issue next chunk's H^T tile into the other buffer (background copy;
        // consumed after next iteration's WAIT_ASYNC + barrier)
        if (n + 1 < NCHUNK) {
            bf16_t* l = &sHT[buf ^ 1][oo * SSTRIDE + jh];
            ASYNC_B128(hsrc + j0 + 32, l, 0);
            ASYNC_B128(hsrc + j0 + 32, l, 16);
        }
#endif
    }

    // --- rowsum reduction (ds_add_f32) ---
    __syncthreads();
    if (t < 32) sRS[t] = 0.0f;
    __syncthreads();
    for (int q = 0; q < 4; ++q) atomicAdd(&sRS[iiq + q], rs[q]);
    __syncthreads();

    // --- normalize + store ---
    for (int r = 0; r < 8; ++r) {
        const int M    = r + (hiH ? 8 : 0);
        const int iloc = rh * 16 + M;
        const float inv = 1.0f / sRS[iloc];
        const int grow  = i0 + iloc;
        const int col   = cg * 32 + l15;
        Z[(size_t)grow * DIM + col]      = acc0[r] * inv;
        Z[(size_t)grow * DIM + col + 16] = acc1[r] * inv;
    }
}

// ---------------------------------------------------------------------------
extern "C" void kernel_launch(void* const* d_in, const int* in_sizes, int n_in,
                              void* d_out, int out_size, void* d_ws, size_t ws_size,
                              hipStream_t stream) {
    (void)in_sizes; (void)n_in; (void)out_size; (void)ws_size;
    const float* x       = (const float*)d_in[0];
    const float* adj     = (const float*)d_in[1];
    const float* w       = (const float*)d_in[2];
    const float* att_src = (const float*)d_in[3];
    const float* att_dst = (const float*)d_in[4];
    float* Z = (float*)d_out;

    char* ws = (char*)d_ws;
    float*  H   = (float*)ws;                               // 4 MB
    bf16_t* HbT = (bf16_t*)(ws + (size_t)4 * 1024 * 1024);  // 2 MB
    float*  es  = (float*)(ws + (size_t)6 * 1024 * 1024);   // 4 x 32 KB
    float*  es2 = es + N_NODES;
    float*  ed  = es + 2 * N_NODES;
    float*  ed2 = es + 3 * N_NODES;

    gat_h_kernel<<<N_NODES / 16, 32, 0, stream>>>(x, w, H, HbT);
    gat_alpha_kernel<<<N_NODES / 256, 256, 0, stream>>>(H, att_src, att_dst,
                                                        es, es2, ed, ed2);
    gat_aggregate_kernel<<<N_NODES / 32, 256, 0, stream>>>(adj, HbT, es, es2,
                                                           ed, ed2, Z);
}
